// LinearAttention_71829033058398
// MI455X (gfx1250) — matrix operand order repaired and checked by reading, not compiled, not run on hardware
//
#include <hip/hip_runtime.h>
#include <math.h>

// ---------------------------------------------------------------------------
// Focused Linear Attention for MI455X (gfx1250), bf16 WMMA pipeline.
// b=4, n=4096, d=1024, h=16, c=64.
// ---------------------------------------------------------------------------

typedef unsigned int u32x4 __attribute__((ext_vector_type(4)));
typedef float        v8f   __attribute__((ext_vector_type(8)));
typedef __bf16       v16bf __attribute__((ext_vector_type(16)));
// TDM descriptor groups (6-arg builtin signature on this toolchain)
typedef unsigned int v4u   __attribute__((ext_vector_type(4)));
typedef int          v8i   __attribute__((ext_vector_type(8)));
typedef int          v4i   __attribute__((ext_vector_type(4)));

union Frag {
  v16bf bf;
  u32x4 q[2];
  unsigned int u[8];
};

__device__ __forceinline__ unsigned short f2bf(float x) {
  unsigned int u = __float_as_uint(x);
  u += 0x7FFFu + ((u >> 16) & 1u);          // round-to-nearest-even
  return (unsigned short)(u >> 16);
}
__device__ __forceinline__ float bf2f(unsigned short u) {
  return __uint_as_float(((unsigned int)u) << 16);
}
__device__ __forceinline__ v8f wmma_bf16(v16bf a, v16bf b, v8f c) {
  return __builtin_amdgcn_wmma_f32_16x16x32_bf16(false, a, false, b,
                                                 (short)0, c, false, false);
}

// Tensor Data Mover: 1-row 2D descriptor copying `nelems` bf16 (contiguous)
// from global to LDS.  D# per CDNA5 ISA ch.8 (group0/group1 bitfields).
__device__ __forceinline__ void tdm_load_bf16(unsigned lds_byte_off,
                                              const void* gaddr,
                                              unsigned nelems) {
  unsigned long long ga = (unsigned long long)gaddr;
  v4u g0;
  g0.x = 1u;                                          // count=1 (valid), user mode
  g0.y = lds_byte_off;                                // lds_addr (bytes)
  g0.z = (unsigned)(ga & 0xFFFFFFFFull);              // global_addr[31:0]
  g0.w = (unsigned)((ga >> 32) & 0x1FFFFFFull)        // global_addr[56:32]
         | (2u << 30);                                // type=2 ("image")
  v8i g1;
  g1[0] = (int)(1u << 16);                 // data_size=1 (2 bytes), wg_mask=0
  g1[1] = (int)((nelems & 0xFFFFu) << 16); // tensor_dim0[15:0]  (bits 63:48)
  g1[2] = (int)(((nelems >> 16) & 0xFFFFu)) | (1 << 16); // dim0 hi, tensor_dim1=1
  g1[3] = (int)((nelems & 0xFFFFu) << 16); // tensor_dim1 hi=0, tile_dim0=nelems
  g1[4] = 1;                               // tile_dim1=1, tile_dim2=0
  g1[5] = (int)nelems;                     // tensor_dim0_stride[31:0]
  g1[6] = 0;                               // stride hi, tensor_dim1_stride lo
  g1[7] = 0;
  v4i z4 = {0, 0, 0, 0};
  v8i z8 = {0, 0, 0, 0, 0, 0, 0, 0};
  __builtin_amdgcn_tensor_load_to_lds(g0, g1, z4, z4, z8, 0);
}

// ---------------------------------------------------------------------------
// Stage A: C[16384,1024] = A[16384,1024] @ W[1024,1024] + bias
// 128x128 block tile, K staged 32 wide through LDS as bf16.
// MODE==0: f32 row-major out.  MODE==1: bf16 per-head-transposed out
//          outT[(b*1024 + col)*4096 + nseq]  (== vT[bh][d'][n]).
// ---------------------------------------------------------------------------
template <int MODE>
__global__ __launch_bounds__(256) void gemm_qkv(
    const float* __restrict__ A, const float* __restrict__ W,
    const float* __restrict__ bias, float* __restrict__ outF,
    unsigned short* __restrict__ outT)
{
  __shared__ __align__(16) unsigned short ldsA[128 * 40]; // [row][k], pad 40
  __shared__ __align__(16) unsigned short ldsB[128 * 40]; // [col][k], pad 40

  const int tid = threadIdx.x, lane = tid & 31, w = tid >> 5, half = lane >> 4;
  const int m0 = blockIdx.y * 128, n0 = blockIdx.x * 128;
  const int wm = w >> 1;   // 0..3 -> 32 rows each
  const int wn = w & 1;    // 0..1 -> 64 cols each

  v8f acc[2][4] = {};

  for (int kt = 0; kt < 32; ++kt) {
    const int k0 = kt * 32;
    __syncthreads();
    // --- load A tile 128x32 (f32 -> bf16), prefetch next K tile ---
#pragma unroll
    for (int it = 0; it < 4; ++it) {
      int idx = it * 256 + tid, base = idx * 4;
      int row = base >> 5, col = base & 31;
      const float* ap = A + (size_t)(m0 + row) * 1024 + k0 + col;
      const float4 v = *(const float4*)ap;
      if (kt < 31) __builtin_prefetch(ap + 32, 0, 0);   // global_prefetch_b8
      unsigned int lo = (unsigned)f2bf(v.x) | ((unsigned)f2bf(v.y) << 16);
      unsigned int hi = (unsigned)f2bf(v.z) | ((unsigned)f2bf(v.w) << 16);
      *(uint2*)(ldsA + row * 40 + col) = make_uint2(lo, hi);
    }
    // --- load W tile 32x128, store transposed [n][k] ---
#pragma unroll
    for (int it = 0; it < 4; ++it) {
      int idx = it * 256 + tid, base = idx * 4;
      int kr = base >> 7, nc = base & 127;
      const float* wp = W + (size_t)(k0 + kr) * 1024 + n0 + nc;
      const float4 v = *(const float4*)wp;
      if (kt < 31) __builtin_prefetch(wp + 32 * 1024, 0, 0);
      ldsB[(nc + 0) * 40 + kr] = f2bf(v.x);
      ldsB[(nc + 1) * 40 + kr] = f2bf(v.y);
      ldsB[(nc + 2) * 40 + kr] = f2bf(v.z);
      ldsB[(nc + 3) * 40 + kr] = f2bf(v.w);
    }
    __syncthreads();

    Frag a[2], b[4];
#pragma unroll
    for (int mt = 0; mt < 2; ++mt) {
      // CDNA5 16-bit A layout: lane L holds M=L%16; K = 8*(L/16) + e (e<8),
      // and K = 16 + 8*(L/16) + (e-8) for e>=8.
      const unsigned short* p = ldsA + (wm * 32 + mt * 16 + (lane & 15)) * 40 + 8 * half;
      a[mt].q[0] = *(const u32x4*)p;
      a[mt].q[1] = *(const u32x4*)(p + 16);
    }
#pragma unroll
    for (int nt = 0; nt < 4; ++nt) {
      // CDNA5 16-bit B layout: lane L holds N=L%16; K = 8*(L/16) + e (e<8), 16 + 8*(L/16) + (e-8) for e>=8.
      const unsigned short* p = ldsB + (wn * 64 + nt * 16 + (lane & 15)) * 40 + 8 * half;
      b[nt].q[0] = *(const u32x4*)p;
      b[nt].q[1] = *(const u32x4*)(p + 16);
    }
#pragma unroll
    for (int mt = 0; mt < 2; ++mt)
#pragma unroll
      for (int nt = 0; nt < 4; ++nt)
        acc[mt][nt] = wmma_bf16(a[mt].bf, b[nt].bf, acc[mt][nt]);
  }

  // --- epilogue: +bias, store (MODE resolved at compile time) ---
#pragma unroll
  for (int mt = 0; mt < 2; ++mt)
#pragma unroll
    for (int nt = 0; nt < 4; ++nt) {
      int gcol = n0 + wn * 64 + nt * 16 + (lane & 15);
      float bv = bias[gcol];
#pragma unroll
      for (int r = 0; r < 8; ++r) {
        int grow = m0 + wm * 32 + mt * 16 + r + 8 * half; // C layout: M=r+8*half
        float val = acc[mt][nt][r] + bv;
        if (MODE == 0) {
          outF[(size_t)grow * 1024 + gcol] = val;
        } else {
          int bb = grow >> 12, nseq = grow & 4095;
          outT[((size_t)(bb * 1024 + gcol)) * 4096 + nseq] = f2bf(val);
        }
      }
    }
}

// ---------------------------------------------------------------------------
// Stage B: focused feature map per row (relu+eps, /softplus(scale),
// cube, norm-preserving rescale over full d=1024).
// TRANSPOSED==0: bf16 row-major.  TRANSPOSED==1: bf16 [bh][c][n].
// ---------------------------------------------------------------------------
template <int TRANSPOSED>
__global__ __launch_bounds__(256) void feature_map(
    const float* __restrict__ X, const float* __restrict__ scale,
    unsigned short* __restrict__ out)
{
  __shared__ float sred[256];
  const int row = blockIdx.x, tid = threadIdx.x;
  const float* xr = X + (size_t)row * 1024;
  const float4 v = *(const float4*)(xr + tid * 4);
  float xin[4] = {v.x, v.y, v.z, v.w};
  float x[4], xp[4];
  float p1 = 0.f;
#pragma unroll
  for (int j = 0; j < 4; ++j) {
    float s = scale[tid * 4 + j];
    float sp = (s > 20.f) ? s : log1pf(expf(s));   // softplus
    x[j] = (fmaxf(xin[j], 0.f) + 1e-6f) / sp;
    p1 += x[j] * x[j];
  }
  sred[tid] = p1; __syncthreads();
  for (int s = 128; s > 0; s >>= 1) { if (tid < s) sred[tid] += sred[tid + s]; __syncthreads(); }
  float n1 = sqrtf(sred[0]); __syncthreads();

  float p2 = 0.f;
#pragma unroll
  for (int j = 0; j < 4; ++j) { xp[j] = x[j] * x[j] * x[j]; p2 += xp[j] * xp[j]; }
  sred[tid] = p2; __syncthreads();
  for (int s = 128; s > 0; s >>= 1) { if (tid < s) sred[tid] += sred[tid + s]; __syncthreads(); }
  float n2 = sqrtf(sred[0]);
  float coef = n1 / fmaxf(n2, 1e-30f);

  if (TRANSPOSED == 0) {
    unsigned int lo = (unsigned)f2bf(xp[0] * coef) | ((unsigned)f2bf(xp[1] * coef) << 16);
    unsigned int hi = (unsigned)f2bf(xp[2] * coef) | ((unsigned)f2bf(xp[3] * coef) << 16);
    *(uint2*)(out + (size_t)row * 1024 + tid * 4) = make_uint2(lo, hi);
  } else {
    int bb = row >> 12, nseq = row & 4095;
#pragma unroll
    for (int j = 0; j < 4; ++j)
      out[((size_t)(bb * 1024 + tid * 4 + j)) * 4096 + nseq] = f2bf(xp[j] * coef);
  }
}

// ---------------------------------------------------------------------------
// Stage C: per (b,h): kv[64,64] = k^T v over K=4096, plus ksum via a
// synthetic "ones column" B fragment (5th N-tile, no loads).
// 8 waves: w&1 = M-half (32 rows), w>>1 = 4-way K split, phased LDS reduce.
// ---------------------------------------------------------------------------
__global__ __launch_bounds__(256) void kv_kernel(
    const unsigned short* __restrict__ kT, const unsigned short* __restrict__ vT,
    unsigned short* __restrict__ kvOut, float* __restrict__ ksum)
{
  __shared__ float red[2][32][80];
  const int bh = blockIdx.x, tid = threadIdx.x, lane = tid & 31, w = tid >> 5;
  const int mh = w & 1, ks = w >> 1, half = lane >> 4;
  const unsigned short* kbase = kT + (size_t)bh * 64 * 4096;
  const unsigned short* vbase = vT + (size_t)bh * 64 * 4096;

  Frag bones;                       // B matrix: column 0 all ones, rest zero
  unsigned short one = ((lane & 15) == 0) ? (unsigned short)0x3F80 : (unsigned short)0;
  unsigned int oo = (unsigned)one | ((unsigned)one << 16);
#pragma unroll
  for (int i = 0; i < 8; ++i) bones.u[i] = oo;

  v8f acc[2][5] = {};
  for (int chunk = ks; chunk < 128; chunk += 4) {
    const int j = chunk * 32;
    Frag a[2], b[4];
#pragma unroll
    for (int mt = 0; mt < 2; ++mt) {
      const unsigned short* p = kbase + (size_t)(mh * 32 + mt * 16 + (lane & 15)) * 4096
                                      + j + 8 * half;
      a[mt].q[0] = *(const u32x4*)p;
      a[mt].q[1] = *(const u32x4*)(p + 16);
    }
#pragma unroll
    for (int nt = 0; nt < 4; ++nt) {
      const unsigned short* p = vbase + (size_t)(nt * 16 + (lane & 15)) * 4096
                                      + j + 8 * half;
      b[nt].q[0] = *(const u32x4*)p;
      b[nt].q[1] = *(const u32x4*)(p + 16);
    }
#pragma unroll
    for (int mt = 0; mt < 2; ++mt) {
#pragma unroll
      for (int nt = 0; nt < 4; ++nt)
        acc[mt][nt] = wmma_bf16(a[mt].bf, b[nt].bf, acc[mt][nt]);
      acc[mt][4] = wmma_bf16(a[mt].bf, bones.bf, acc[mt][4]);   // ksum column
    }
  }

  // deterministic phased reduction over the 4 K-splits
  for (int phase = 0; phase < 4; ++phase) {
    if (ks == phase) {
#pragma unroll
      for (int mt = 0; mt < 2; ++mt)
#pragma unroll
        for (int nt = 0; nt < 5; ++nt)
#pragma unroll
          for (int r = 0; r < 8; ++r) {
            int rowl = mt * 16 + r + 8 * half;
            int col = nt * 16 + (lane & 15);
            if (phase == 0) red[mh][rowl][col]  = acc[mt][nt][r];
            else            red[mh][rowl][col] += acc[mt][nt][r];
          }
    }
    __syncthreads();
  }

  for (int idx = tid; idx < 64 * 80; idx += 256) {
    int rr = idx / 80, col = idx % 80;
    float vsum = red[rr >> 5][rr & 31][col];
    if (col < 64)       kvOut[(size_t)bh * 4096 + rr * 64 + col] = f2bf(vsum);
    else if (col == 64) ksum[bh * 64 + rr] = vsum;
  }
}

// ---------------------------------------------------------------------------
// Stage D: per (b,h, 64-row i-tile): x = (q @ kv) * z,  z = 1/(q.ksum + eps).
// kv tile (8 KB contiguous bf16) fetched via Tensor Data Mover into LDS.
// ---------------------------------------------------------------------------
__global__ __launch_bounds__(256) void attn_out(
    const unsigned short* __restrict__ qBF, const unsigned short* __restrict__ kvBF,
    const float* __restrict__ ksum, float* __restrict__ out)
{
  __shared__ __align__(16) unsigned short kvRaw[4096];  // [c][d'] as stored
  __shared__ __align__(16) unsigned short kvT[64 * 72]; // [d'][c], pad 72
  __shared__ float zbuf[64];
  __shared__ float ksums[64];

  const int bh = blockIdx.y, b = bh >> 4, h = bh & 15;
  const int i0 = blockIdx.x * 64;
  const int tid = threadIdx.x, lane = tid & 31, w = tid >> 5, half = lane >> 4;

  // TDM: async DMA of kv[bh] (4096 bf16) into LDS; issued once (EXEC-agnostic).
  if (tid == 0) {
    tdm_load_bf16((unsigned)(unsigned long long)(const void*)kvRaw,
                  kvBF + (size_t)bh * 4096, 4096);
  }
  if (tid < 64) ksums[tid] = ksum[bh * 64 + tid];
  __builtin_amdgcn_s_wait_tensorcnt(0);   // wave0 waits for DMA; others: no-op
  __syncthreads();

  // transpose kvRaw [c][d'] -> kvT [d'][c]
  for (int e = tid; e < 4096; e += 256) {
    int c = e >> 6, dp = e & 63;
    kvT[dp * 72 + c] = kvRaw[e];
  }
  __syncthreads();

  if (tid < 64) {
    const unsigned short* qr = qBF + ((size_t)(b * 4096 + i0 + tid)) * 1024 + h * 64;
    float d = 0.f;
    for (int c = 0; c < 64; ++c) d += bf2f(qr[c]) * ksums[c];
    zbuf[tid] = 1.0f / (d + 1e-6f);
  }
  __syncthreads();

  const int wm = w >> 1;  // 4 M-tiles of 16 rows
  const int wn = w & 1;   // 2 halves of 32 cols
  v8f acc[2] = {};
#pragma unroll
  for (int kc = 0; kc < 64; kc += 32) {
    Frag a, bf[2];
    const unsigned short* p = qBF + ((size_t)(b * 4096 + i0 + wm * 16 + (lane & 15))) * 1024
                                  + h * 64 + kc + 8 * half;
    a.q[0] = *(const u32x4*)p;
    a.q[1] = *(const u32x4*)(p + 16);
#pragma unroll
    for (int nt = 0; nt < 2; ++nt) {
      const unsigned short* q = kvT + (wn * 32 + nt * 16 + (lane & 15)) * 72 + kc + 8 * half;
      bf[nt].q[0] = *(const u32x4*)q;
      bf[nt].q[1] = *(const u32x4*)(q + 16);
    }
#pragma unroll
    for (int nt = 0; nt < 2; ++nt)
      acc[nt] = wmma_bf16(a.bf, bf[nt].bf, acc[nt]);
  }
#pragma unroll
  for (int nt = 0; nt < 2; ++nt)
#pragma unroll
    for (int r = 0; r < 8; ++r) {
      int rowl = wm * 16 + r + 8 * half;
      int col = wn * 32 + nt * 16 + (lane & 15);
      out[((size_t)(b * 4096 + i0 + rowl)) * 1024 + h * 64 + col] = acc[nt][r] * zbuf[rowl];
    }
}

// ---------------------------------------------------------------------------
extern "C" void kernel_launch(void* const* d_in, const int* in_sizes, int n_in,
                              void* d_out, int out_size, void* d_ws, size_t ws_size,
                              hipStream_t stream)
{
  const float* in_q  = (const float*)d_in[0];
  const float* in_k  = (const float*)d_in[1];
  const float* in_v  = (const float*)d_in[2];
  const float* Wq    = (const float*)d_in[3];
  const float* bq    = (const float*)d_in[4];
  const float* Wk    = (const float*)d_in[5];
  const float* bk    = (const float*)d_in[6];
  const float* Wv    = (const float*)d_in[7];
  const float* bv    = (const float*)d_in[8];
  const float* scale = (const float*)d_in[9];

  char* ws = (char*)d_ws;
  const size_t MB = 1024 * 1024;
  float*          kF   = (float*)(ws);                         // 64 MB  k proj f32
  unsigned short* qBF  = (unsigned short*)(ws + 64 * MB);      // 32 MB  q feat bf16
  unsigned short* kT   = (unsigned short*)(ws + 96 * MB);      // 32 MB  k feat bf16 [bh][c][n]
  unsigned short* vT   = (unsigned short*)(ws + 128 * MB);     // 32 MB  v proj bf16 [bh][d'][n]
  unsigned short* kvB  = (unsigned short*)(ws + 160 * MB);     // 512 KB kv bf16
  float*          ksum = (float*)(ws + 160 * MB + 512 * 1024); // 16 KB
  float*          qF   = (float*)d_out;                        // q proj f32 reuses d_out

  dim3 blk(256);
  dim3 gGemm(8, 128);
  gemm_qkv<0><<<gGemm, blk, 0, stream>>>(in_q, Wq, bq, qF, nullptr);
  gemm_qkv<0><<<gGemm, blk, 0, stream>>>(in_k, Wk, bk, kF, nullptr);
  gemm_qkv<1><<<gGemm, blk, 0, stream>>>(in_v, Wv, bv, nullptr, vT);
  feature_map<0><<<16384, blk, 0, stream>>>(qF, scale, qBF);
  feature_map<1><<<16384, blk, 0, stream>>>(kF, scale, kT);
  kv_kernel<<<64, blk, 0, stream>>>(kT, vT, kvB, ksum);
  attn_out<<<dim3(64, 64), blk, 0, stream>>>(qBF, kvB, ksum, (float*)d_out);
}
